// PropNetDiffDenModel_88304527606635
// MI455X (gfx1250) — matrix operand order repaired and checked by reading, not compiled, not run on hardware
//
#include <hip/hip_runtime.h>
#include <hip/hip_bf16.h>
#include <cstdint>

typedef _Float16 half_t;
typedef __attribute__((ext_vector_type(8)))  _Float16 v8h;
typedef __attribute__((ext_vector_type(16))) _Float16 v16h;
typedef __attribute__((ext_vector_type(8)))  float    v8f;

#define BB 4
#define NHH 4
#define NN 1000
#define NF 256
#define TK 10
#define RR (NN*TK)     // 10000 edges per batch
#define BN (BB*NN)     // 4000 nodes total
#define BR (BB*RR)     // 40000 edges total
#define BIGD 1e10f
#define THR2 0.25f     // ADJ_THRESH^2

__device__ __forceinline__ bool better(float d1, int j1, float d2, int j2) {
  return (d1 < d2) || (d1 == d2 && j1 < j2);
}

// ---------------- weight pack: W (K x 256) f32 -> Wt (256 x K) f16 ----------
__global__ void packWt_kernel(const float* __restrict__ W, half_t* __restrict__ Wt, int K) {
  int idx = blockIdx.x * blockDim.x + threadIdx.x;
  if (idx >= K * NF) return;
  int k = idx % K;
  int n = idx / K;
  Wt[(size_t)n * K + k] = (half_t)W[(size_t)k * NF + n];
}

// ---------------- input prep: a_cur, s_cur, pe_in ---------------------------
__global__ void prep_kernel(const float* __restrict__ a_hist, const float* __restrict__ s_hist,
                            const float* __restrict__ s_delta,
                            float* __restrict__ a_cur, float* __restrict__ s_cur,
                            float* __restrict__ pe_in) {
  int t = blockIdx.x * blockDim.x + threadIdx.x;
  if (t >= BN) return;
  int b = t / NN, n = t - b * NN;
  a_cur[t] = a_hist[(b * NHH + NHH - 1) * NN + n];
#pragma unroll
  for (int d = 0; d < 3; ++d)
    s_cur[t * 3 + d] = s_hist[((b * NHH + NHH - 1) * NN + n) * 3 + d];
#pragma unroll
  for (int h = 0; h < NHH; ++h) {
#pragma unroll
    for (int d = 0; d < 3; ++d)
      pe_in[t * 16 + h * 3 + d] = s_delta[((b * NHH + h) * NN + n) * 3 + d];
    pe_in[t * 16 + 12 + h] = a_hist[(b * NHH + h) * NN + n];
  }
}

// ---------------- top-k edge construction (one wave per node) ---------------
__global__ __launch_bounds__(32)
void topk_kernel(const float* __restrict__ s_cur, const float* __restrict__ a_cur,
                 int* __restrict__ sendIdx, float* __restrict__ flag) {
  __shared__ float sx[NN], sy[NN], sz[NN];
  __shared__ unsigned char tl[NN];
  __shared__ float mv[32 * TK];
  __shared__ int   mi[32 * TK];
  __shared__ int   hd[32];
  const int b = blockIdx.x / NN;
  const int i = blockIdx.x % NN;
  const int lane = threadIdx.x;
  for (int j = lane; j < NN; j += 32) {
    int g = b * NN + j;
    sx[j] = s_cur[g * 3 + 0];
    sy[j] = s_cur[g * 3 + 1];
    sz[j] = s_cur[g * 3 + 2];
    tl[j] = a_cur[g] > 0.5f ? 1 : 0;
  }
  hd[lane] = 0;
  __syncthreads();
  const float px = sx[i], py = sy[i], pz = sz[i];
  const int ti = tl[i];
  float bv[TK]; int bj[TK];
#pragma unroll
  for (int p = 0; p < TK; ++p) { bv[p] = 3.0e38f; bj[p] = 0x7fffffff; }
  for (int j = lane; j < NN; j += 32) {
    float dx = px - sx[j], dy = py - sy[j], dz = pz - sz[j];
    float d = dx * dx + dy * dy + dz * dz;
    if (ti && tl[j]) d = BIGD;
    float cd = d; int cj = j;
#pragma unroll
    for (int p = 0; p < TK; ++p) {   // bubble-insert, keeps list sorted by (d, idx)
      if (better(cd, cj, bv[p], bj[p])) {
        float tv = bv[p]; int tj = bj[p];
        bv[p] = cd; bj[p] = cj; cd = tv; cj = tj;
      }
    }
  }
#pragma unroll
  for (int p = 0; p < TK; ++p) { mv[lane * TK + p] = bv[p]; mi[lane * TK + p] = bj[p]; }
  __syncthreads();
  if (lane == 0) {
    for (int k = 0; k < TK; ++k) {
      float bestv = 3.0e38f; int bestj = 0x7fffffff; int bt = -1;
      for (int t = 0; t < 32; ++t) {
        int h = hd[t];
        if (h < TK) {
          float v = mv[t * TK + h]; int jj = mi[t * TK + h];
          if (bt < 0 || better(v, jj, bestv, bestj)) { bestv = v; bestj = jj; bt = t; }
        }
      }
      hd[bt]++;
      int er = b * RR + i * TK + k;
      sendIdx[er] = bestj;
      flag[er] = (bestv < THR2 && !ti) ? 1.0f : 0.0f;
    }
  }
}

// ---------------- small VALU layers -----------------------------------------
__global__ void pe0_kernel(const float* __restrict__ pe_in, const float* __restrict__ w0,
                           const float* __restrict__ b0, half_t* __restrict__ out) {
  int idx = blockIdx.x * blockDim.x + threadIdx.x;
  if (idx >= BN * NF) return;
  int r = idx >> 8, c = idx & 255;
  float acc = b0[c];
#pragma unroll
  for (int k = 0; k < 16; ++k) acc += pe_in[r * 16 + k] * w0[k * NF + c];
  out[idx] = (half_t)fmaxf(acc, 0.0f);
}

__global__ void re0_kernel(const float* __restrict__ a_cur, const float* __restrict__ s_cur,
                           const int* __restrict__ sendIdx, const float* __restrict__ flag,
                           const float* __restrict__ w0, const float* __restrict__ b0,
                           half_t* __restrict__ out) {
  int idx = blockIdx.x * blockDim.x + threadIdx.x;
  if (idx >= BR * NF) return;
  int r = idx >> 8, c = idx & 255;
  int b = r / RR; int rr = r - b * RR;
  int ig = b * NN + rr / TK;
  int sg = b * NN + sendIdx[r];
  float f = flag[r];
  float x0 = f * a_cur[ig];
  float x1 = f * a_cur[sg];
  float x2 = f * (s_cur[ig * 3 + 0] - s_cur[sg * 3 + 0]);
  float x3 = f * (s_cur[ig * 3 + 1] - s_cur[sg * 3 + 1]);
  float x4 = f * (s_cur[ig * 3 + 2] - s_cur[sg * 3 + 2]);
  float acc = b0[c] + x0 * w0[c] + x1 * w0[NF + c] + x2 * w0[2 * NF + c]
            + x3 * w0[3 * NF + c] + x4 * w0[4 * NF + c];
  out[idx] = (half_t)fmaxf(acc, 0.0f);
}

// agg[n,c] = sum_k flag * effect_rel[n*10+k, c]  -> ppin cols 256..511
__global__ void agg_kernel(const float* __restrict__ flag, const half_t* __restrict__ effrel,
                           half_t* __restrict__ ppin) {
  int idx = blockIdx.x * blockDim.x + threadIdx.x;
  if (idx >= BN * NF) return;
  int ng = idx >> 8, c = idx & 255;
  int b = ng / NN; int n = ng - b * NN;
  int e0 = b * RR + n * TK;
  float s = 0.0f;
#pragma unroll
  for (int k = 0; k < TK; ++k)
    s += flag[e0 + k] * (float)effrel[(size_t)(e0 + k) * NF + c];
  ppin[(size_t)ng * 512 + 256 + c] = (half_t)s;
}

__global__ void copy_pe_kernel(const half_t* __restrict__ peH, half_t* __restrict__ ppin) {
  int idx = blockIdx.x * blockDim.x + threadIdx.x;
  if (idx >= BN * NF) return;
  int r = idx >> 8, c = idx & 255;
  ppin[(size_t)r * 512 + c] = peH[idx];
}

__global__ void final_kernel(const half_t* __restrict__ hid, const float* __restrict__ w1,
                             const float* __restrict__ b1, const float* __restrict__ s_cur,
                             float* __restrict__ out) {
  int idx = blockIdx.x * blockDim.x + threadIdx.x;
  if (idx >= BN * 3) return;
  int r = idx / 3, d = idx - r * 3;
  float acc = b1[d];
  for (int c = 0; c < NF; ++c) acc += (float)hid[(size_t)r * NF + c] * w1[c * 3 + d];
  out[idx] = acc + s_cur[r * 3 + d];
}

// ---------------- generic WMMA GEMM: out = relu(A @ W + bias [+ resid]) -----
// A: (M x K) f16 row-major, Wt: (256 x K) f16 (= W transposed). 4 waves/block.
// Each wave computes a 32x64 strip: 2 M-subtiles share every B fragment, so
// per k-step: 2 A-loads + 4 B-loads feed 8 v_wmma (ratio 0.75 loads/wmma).
__global__ __launch_bounds__(128)
void gemm16_kernel(const half_t* __restrict__ A, int lda,
                   const half_t* __restrict__ Wt,
                   const float* __restrict__ bias,
                   const float* resid,      // may alias outF (no restrict)
                   float* outF,             // nullable
                   half_t* __restrict__ outH, // nullable
                   int K) {
  const int lane = threadIdx.x & 31;
  const int wv   = threadIdx.x >> 5;
  const int m0   = blockIdx.x * 32;
  const int r16  = lane & 15;
  const int hi   = lane >> 4;            // lane half selects K sub-range
  const int n0   = wv * 64;
  v8f acc[2][4] = {};
  const half_t* arow0 = A + (size_t)(m0 + r16) * lda;
  const half_t* arow1 = A + (size_t)(m0 + 16 + r16) * lda;
  for (int k0 = 0; k0 < K; k0 += 32) {
    const int ca = k0 + hi * 8;
    v8h a0lo = *(const v8h*)(arow0 + ca);
    v8h a0hi = *(const v8h*)(arow0 + ca + 16);
    v16h af0 = __builtin_shufflevector(a0lo, a0hi, 0,1,2,3,4,5,6,7,8,9,10,11,12,13,14,15);
    v8h a1lo = *(const v8h*)(arow1 + ca);
    v8h a1hi = *(const v8h*)(arow1 + ca + 16);
    v16h af1 = __builtin_shufflevector(a1lo, a1hi, 0,1,2,3,4,5,6,7,8,9,10,11,12,13,14,15);
    const int kb = k0 + hi * 8;
#pragma unroll
    for (int t = 0; t < 4; ++t) {
      const half_t* wp = Wt + (size_t)(n0 + 16 * t + r16) * K + kb;
      v8h blo = *(const v8h*)wp;
      v8h bhi = *(const v8h*)(wp + 16);
      v16h bf = __builtin_shufflevector(blo, bhi, 0,1,2,3,4,5,6,7,8,9,10,11,12,13,14,15);
      acc[0][t] = __builtin_amdgcn_wmma_f32_16x16x32_f16(false, af0, false, bf,
                                                         (short)0, acc[0][t], false, false);
      acc[1][t] = __builtin_amdgcn_wmma_f32_16x16x32_f16(false, af1, false, bf,
                                                         (short)0, acc[1][t], false, false);
    }
  }
#pragma unroll
  for (int mm = 0; mm < 2; ++mm) {
#pragma unroll
    for (int t = 0; t < 4; ++t) {
      const int col = n0 + 16 * t + r16;
      const float bs = bias[col];
#pragma unroll
      for (int v = 0; v < 8; ++v) {
        const int r = m0 + 16 * mm + hi * 8 + v;
        float val = acc[mm][t][v] + bs;
        if (resid) val += resid[(size_t)r * NF + col];
        val = fmaxf(val, 0.0f);
        if (outF) outF[(size_t)r * NF + col] = val;
        if (outH) outH[(size_t)r * NF + col] = (half_t)val;
      }
    }
  }
}

// ---------------- fused relation-propagation GEMM (K=768) -------------------
// A row r = [relation_encode[r] | flag*pe[recv(r)] | flag*pe[send(r)]],
// gathered on the fly in the A-fragment loader (no 61MB concat buffer).
// Same 32x64 per-wave tiling as gemm16_kernel.
__global__ __launch_bounds__(128)
void rp_gemm_kernel(const half_t* __restrict__ relE, const half_t* __restrict__ peH,
                    const int* __restrict__ sendIdx, const float* __restrict__ flag,
                    const half_t* __restrict__ Wt, const float* __restrict__ bias,
                    half_t* __restrict__ outH) {
  const int lane = threadIdx.x & 31;
  const int wv   = threadIdx.x >> 5;
  const int m0   = blockIdx.x * 32;
  const int r16  = lane & 15;
  const int hi   = lane >> 4;
  const int n0   = wv * 64;

  const int row0 = m0 + r16;
  const int row1 = m0 + 16 + r16;
  const int b0_  = row0 / RR;
  const int b1_  = row1 / RR;
  const int rr0  = row0 - b0_ * RR;
  const int rr1  = row1 - b1_ * RR;
  const bool f0 = flag[row0] > 0.5f;
  const bool f1 = flag[row1] > 0.5f;
  const half_t* rel0 = relE + (size_t)row0 * NF;
  const half_t* rel1 = relE + (size_t)row1 * NF;
  const half_t* rcv0 = peH + (size_t)(b0_ * NN + rr0 / TK) * NF;
  const half_t* rcv1 = peH + (size_t)(b1_ * NN + rr1 / TK) * NF;
  const half_t* snd0 = peH + (size_t)(b0_ * NN + sendIdx[row0]) * NF;
  const half_t* snd1 = peH + (size_t)(b1_ * NN + sendIdx[row1]) * NF;

  v8f acc[2][4] = {};
  const v8h zero = {};
  for (int k0 = 0; k0 < 768; k0 += 32) {   // k0 uniform -> section branch uniform
    const int c0 = k0 + hi * 8;
    v8h a0lo, a0hi, a1lo, a1hi;
    if (k0 < 256) {
      a0lo = *(const v8h*)(rel0 + c0);        a0hi = *(const v8h*)(rel0 + c0 + 16);
      a1lo = *(const v8h*)(rel1 + c0);        a1hi = *(const v8h*)(rel1 + c0 + 16);
    } else if (k0 < 512) {
      const int c = c0 - 256;
      a0lo = *(const v8h*)(rcv0 + c);         a0hi = *(const v8h*)(rcv0 + c + 16);
      a1lo = *(const v8h*)(rcv1 + c);         a1hi = *(const v8h*)(rcv1 + c + 16);
      if (!f0) { a0lo = zero; a0hi = zero; }
      if (!f1) { a1lo = zero; a1hi = zero; }
    } else {
      const int c = c0 - 512;
      a0lo = *(const v8h*)(snd0 + c);         a0hi = *(const v8h*)(snd0 + c + 16);
      a1lo = *(const v8h*)(snd1 + c);         a1hi = *(const v8h*)(snd1 + c + 16);
      if (!f0) { a0lo = zero; a0hi = zero; }
      if (!f1) { a1lo = zero; a1hi = zero; }
    }
    v16h af0 = __builtin_shufflevector(a0lo, a0hi, 0,1,2,3,4,5,6,7,8,9,10,11,12,13,14,15);
    v16h af1 = __builtin_shufflevector(a1lo, a1hi, 0,1,2,3,4,5,6,7,8,9,10,11,12,13,14,15);
    const int kb = k0 + hi * 8;
#pragma unroll
    for (int t = 0; t < 4; ++t) {
      const half_t* wp = Wt + (size_t)(n0 + 16 * t + r16) * 768 + kb;
      v8h blo = *(const v8h*)wp;
      v8h bhi = *(const v8h*)(wp + 16);
      v16h bf = __builtin_shufflevector(blo, bhi, 0,1,2,3,4,5,6,7,8,9,10,11,12,13,14,15);
      acc[0][t] = __builtin_amdgcn_wmma_f32_16x16x32_f16(false, af0, false, bf,
                                                         (short)0, acc[0][t], false, false);
      acc[1][t] = __builtin_amdgcn_wmma_f32_16x16x32_f16(false, af1, false, bf,
                                                         (short)0, acc[1][t], false, false);
    }
  }
#pragma unroll
  for (int mm = 0; mm < 2; ++mm) {
#pragma unroll
    for (int t = 0; t < 4; ++t) {
      const int col = n0 + 16 * t + r16;
      const float bs = bias[col];
#pragma unroll
      for (int v = 0; v < 8; ++v) {
        const int r = m0 + 16 * mm + hi * 8 + v;
        outH[(size_t)r * NF + col] = (half_t)fmaxf(acc[mm][t][v] + bs, 0.0f);
      }
    }
  }
}

// ---------------- host orchestration ----------------------------------------
extern "C" void kernel_launch(void* const* d_in, const int* in_sizes, int n_in,
                              void* d_out, int out_size, void* d_ws, size_t ws_size,
                              hipStream_t stream) {
  (void)in_sizes; (void)n_in; (void)out_size; (void)ws_size;
  const float* a_hist = (const float*)d_in[0];
  const float* s_hist = (const float*)d_in[1];
  const float* s_delta = (const float*)d_in[2];
  const float* pe_w0 = (const float*)d_in[3];
  const float* pe_b0 = (const float*)d_in[4];
  const float* pe_w1 = (const float*)d_in[5];
  const float* pe_b1 = (const float*)d_in[6];
  const float* re_w0 = (const float*)d_in[7];
  const float* re_b0 = (const float*)d_in[8];
  const float* re_w1 = (const float*)d_in[9];
  const float* re_b1 = (const float*)d_in[10];
  const float* re_w2 = (const float*)d_in[11];
  const float* re_b2 = (const float*)d_in[12];
  const float* rp_w  = (const float*)d_in[13];
  const float* rp_b  = (const float*)d_in[14];
  const float* pp_w  = (const float*)d_in[15];
  const float* pp_b  = (const float*)d_in[16];
  const float* pr_w0 = (const float*)d_in[17];
  const float* pr_b0 = (const float*)d_in[18];
  const float* pr_w1 = (const float*)d_in[19];
  const float* pr_b1 = (const float*)d_in[20];
  float* out = (float*)d_out;

  char* ws = (char*)d_ws;
  size_t off = 0;
  auto alloc = [&](size_t bytes) -> void* {
    off = (off + 255) & ~(size_t)255;
    void* p = ws + off;
    off += bytes;
    return p;
  };
  half_t* reA     = (half_t*)alloc((size_t)BR * NF * 2);   // relation hidden / encode
  half_t* reB     = (half_t*)alloc((size_t)BR * NF * 2);   // relation hidden / effect_rel
  half_t* ppin    = (half_t*)alloc((size_t)BN * 512 * 2);  // [particle_encode | agg]
  half_t* peH0    = (half_t*)alloc((size_t)BN * NF * 2);   // particle layer0 out
  half_t* peF16   = (half_t*)alloc((size_t)BN * NF * 2);   // particle_effect f16
  float*  peF32   = (float*) alloc((size_t)BN * NF * 4);   // particle_effect f32 (residual)
  half_t* prHid   = (half_t*)alloc((size_t)BN * NF * 2);
  float*  pe_in   = (float*) alloc((size_t)BN * 16 * 4);
  float*  a_cur   = (float*) alloc((size_t)BN * 4);
  float*  s_cur   = (float*) alloc((size_t)BN * 3 * 4);
  int*    sendIdx = (int*)   alloc((size_t)BR * 4);
  float*  flag    = (float*) alloc((size_t)BR * 4);
  half_t* pe_w1t  = (half_t*)alloc((size_t)NF * 256 * 2);
  half_t* re_w1t  = (half_t*)alloc((size_t)NF * 256 * 2);
  half_t* re_w2t  = (half_t*)alloc((size_t)NF * 256 * 2);
  half_t* rp_wt   = (half_t*)alloc((size_t)NF * 768 * 2);
  half_t* pp_wt   = (half_t*)alloc((size_t)NF * 512 * 2);
  half_t* pr_w0t  = (half_t*)alloc((size_t)NF * 256 * 2);

  // pack weights transposed -> f16
  packWt_kernel<<<(256 * NF + 255) / 256, 256, 0, stream>>>(pe_w1, pe_w1t, 256);
  packWt_kernel<<<(256 * NF + 255) / 256, 256, 0, stream>>>(re_w1, re_w1t, 256);
  packWt_kernel<<<(256 * NF + 255) / 256, 256, 0, stream>>>(re_w2, re_w2t, 256);
  packWt_kernel<<<(768 * NF + 255) / 256, 256, 0, stream>>>(rp_w,  rp_wt,  768);
  packWt_kernel<<<(512 * NF + 255) / 256, 256, 0, stream>>>(pp_w,  pp_wt,  512);
  packWt_kernel<<<(256 * NF + 255) / 256, 256, 0, stream>>>(pr_w0, pr_w0t, 256);

  prep_kernel<<<(BN + 127) / 128, 128, 0, stream>>>(a_hist, s_hist, s_delta, a_cur, s_cur, pe_in);
  topk_kernel<<<BN, 32, 0, stream>>>(s_cur, a_cur, sendIdx, flag);

  // particle encoder
  pe0_kernel<<<(BN * NF + 255) / 256, 256, 0, stream>>>(pe_in, pe_w0, pe_b0, peH0);
  gemm16_kernel<<<BN / 32, 128, 0, stream>>>(peH0, NF, pe_w1t, pe_b1, nullptr, peF32, peF16, 256);
  copy_pe_kernel<<<(BN * NF + 255) / 256, 256, 0, stream>>>(peF16, ppin);

  // relation encoder
  re0_kernel<<<(BR * NF + 255) / 256, 256, 0, stream>>>(a_cur, s_cur, sendIdx, flag, re_w0, re_b0, reA);
  gemm16_kernel<<<BR / 32, 128, 0, stream>>>(reA, NF, re_w1t, re_b1, nullptr, nullptr, reB, 256);
  gemm16_kernel<<<BR / 32, 128, 0, stream>>>(reB, NF, re_w2t, re_b2, nullptr, nullptr, reA, 256);

  // propagation: 3 steps
  for (int it = 0; it < 3; ++it) {
    rp_gemm_kernel<<<BR / 32, 128, 0, stream>>>(reA, peF16, sendIdx, flag, rp_wt, rp_b, reB);
    agg_kernel<<<(BN * NF + 255) / 256, 256, 0, stream>>>(flag, reB, ppin);
    gemm16_kernel<<<BN / 32, 128, 0, stream>>>(ppin, 512, pp_wt, pp_b, peF32, peF32, peF16, 512);
  }

  // predictor
  gemm16_kernel<<<BN / 32, 128, 0, stream>>>(peF16, NF, pr_w0t, pr_b0, nullptr, nullptr, prHid, 256);
  final_kernel<<<(BN * 3 + 255) / 256, 256, 0, stream>>>(prHid, pr_w1, pr_b1, s_cur, out);
}
